// ForceField_77627238908556
// MI455X (gfx1250) — matrix operand order repaired and checked by reading, not compiled, not run on hardware
//
#include <hip/hip_runtime.h>
#include <hip/hip_bf16.h>

typedef __attribute__((ext_vector_type(16))) _Float16 v16h;
typedef __attribute__((ext_vector_type(8)))  float    v8f;

#define LN 128   // ligand atoms
#define RN 1024  // receptor atoms
#define EN 64    // RBF steps
#define FN 64    // feature size
#define PN 8     // poses

#define ECHUNK 16                 // e-values per wave
#define NWAVE  (512 * (EN/ECHUNK))// 512 tiles * 4 e-chunks = 2048 waves

// sqrt(40.96 * log2(e)) : pre-scales distance so rbf = exp2(-(d' - mu')^2)
#define SQ_C 7.6871841f
#define MU_STEP (10.0f / 63.0f)   // jnp.linspace(0, 10, 64)

// load 8 consecutive f32 and place as f16 into frag[base..base+7]
__device__ __forceinline__ void load8h(const float* __restrict__ src, v16h& dst, int base) {
    const float4 a = *reinterpret_cast<const float4*>(src);
    const float4 b = *reinterpret_cast<const float4*>(src + 4);
    dst[base+0] = (_Float16)a.x; dst[base+1] = (_Float16)a.y;
    dst[base+2] = (_Float16)a.z; dst[base+3] = (_Float16)a.w;
    dst[base+4] = (_Float16)b.x; dst[base+5] = (_Float16)b.y;
    dst[base+6] = (_Float16)b.z; dst[base+7] = (_Float16)b.w;
}

__global__ __launch_bounds__(256) void
ff_fused_kernel(const float* __restrict__ lig_feat,   // [L][E][F]
                const float* __restrict__ rec_feat,   // [R][E][F]
                const float* __restrict__ lig_coords, // [P][L][3]
                const float* __restrict__ rec_coords, // [R][3]
                float* __restrict__ partials)         // [NWAVE][P]
{
    const int lane = threadIdx.x & 31;
    const int wave = threadIdx.x >> 5;
    const int w    = blockIdx.x * 8 + wave;      // 0..2047
    const int tile = w >> 2;                     // 0..511
    const int ec   = w & 3;                      // e-chunk 0..3
    const int tl   = tile >> 6;                  // 0..7   (L/16)
    const int tr   = tile & 63;                  // 0..63  (R/16)
    const int l0   = tl * 16, r0 = tr * 16;
    const int half = lane >> 4;                  // wave32 half
    const int lc   = lane & 15;

    // ---- pre-scaled distances for the C-matrix positions this lane owns ----
    // C layout: VGPR v -> M = v + 8*half, lane -> N = lc
    const float* rc = rec_coords + (size_t)(r0 + lc) * 3;
    const float rx = rc[0], ry = rc[1], rz = rc[2];
    float d[PN * 8];
#pragma unroll
    for (int v = 0; v < 8; ++v) {
        const int lrow = l0 + v + 8 * half;
#pragma unroll
        for (int p = 0; p < PN; ++p) {
            const float* lcrd = lig_coords + ((size_t)p * LN + lrow) * 3;
            const float dx = lcrd[0] - rx, dy = lcrd[1] - ry, dz = lcrd[2] - rz;
            d[p * 8 + v] = sqrtf(dx * dx + dy * dy + dz * dz) * SQ_C;
        }
    }

    float sums[PN];
#pragma unroll
    for (int p = 0; p < PN; ++p) sums[p] = 0.0f;

    // rows this lane loads for A/B fragments (both use M/N = lane&15)
    const float* Arow = lig_feat + (size_t)(l0 + lc) * EN * FN;
    const float* Brow = rec_feat + (size_t)(r0 + lc) * EN * FN;

    const int e0 = ec * ECHUNK;
    for (int e = e0; e < e0 + ECHUNK; ++e) {
        const float* Ae = Arow + (size_t)e * FN;
        const float* Be = Brow + (size_t)e * FN;

        v8f acc = {};  // atn tile accumulator (f32)
#pragma unroll
        for (int c = 0; c < 2; ++c) {        // two K=32 chunks cover F=64
            v16h a, b;
            // A frag: i<8 -> f = 32c + 8h + i ; i>=8 -> f = 32c + 16 + 8h + (i-8)
            load8h(Ae + 32 * c + 8 * half,      a, 0);
            load8h(Ae + 32 * c + 16 + 8 * half, a, 8);
            // B frag: i<8 -> f = 32c + 8h + i ; i>=8 -> f = 32c + 16 + 8h + (i-8)
            load8h(Be + 32 * c + 8 * half,      b, 0);
            load8h(Be + 32 * c + 16 + 8 * half, b, 8);
            acc = __builtin_amdgcn_wmma_f32_16x16x32_f16(
                false, a, false, b, (short)0, acc, false, false);
        }

        // consume atn tile: sums[p] += exp2(-(d' - mu')^2) * atn
        const float mus = (MU_STEP * SQ_C) * (float)e;
#pragma unroll
        for (int v = 0; v < 8; ++v) {
            const float av = acc[v];
#pragma unroll
            for (int p = 0; p < PN; ++p) {
                const float t = d[p * 8 + v] - mus;
                // raw v_exp_f32: arg <= 0, large-negative underflows to 0 natively
                sums[p] = fmaf(__builtin_amdgcn_exp2f(-t * t), av, sums[p]);
            }
        }
    }

    // wave32 butterfly reduction, lane 0 writes the 8 partials
#pragma unroll
    for (int p = 0; p < PN; ++p) {
        float s = sums[p];
#pragma unroll
        for (int off = 16; off > 0; off >>= 1)
            s += __shfl_xor(s, off, 32);
        if (lane == 0) partials[(size_t)w * PN + p] = s;
    }
}

// deterministic final reduction: fixed summation order per pose
__global__ void ff_reduce_kernel(const float* __restrict__ partials,
                                 float* __restrict__ out)
{
    const int p = threadIdx.x;   // 8 threads
    if (p >= PN) return;
    float s = 0.0f;
    for (int t = 0; t < NWAVE; ++t)
        s += partials[(size_t)t * PN + p];
    out[p] = s * 0.1f;           // ENERGY_SCALE
}

extern "C" void kernel_launch(void* const* d_in, const int* in_sizes, int n_in,
                              void* d_out, int out_size, void* d_ws, size_t ws_size,
                              hipStream_t stream) {
    const float* lig_feat   = (const float*)d_in[0];
    const float* rec_feat   = (const float*)d_in[1];
    const float* lig_coords = (const float*)d_in[2];
    const float* rec_coords = (const float*)d_in[3];
    float* partials = (float*)d_ws;   // NWAVE * PN * 4 = 64 KB

    ff_fused_kernel<<<NWAVE / 8, 256, 0, stream>>>(
        lig_feat, rec_feat, lig_coords, rec_coords, partials);
    ff_reduce_kernel<<<1, PN, 0, stream>>>(partials, (float*)d_out);
}
